// KVR_Spatial_SelfAttention_18133351923890
// MI455X (gfx1250) — matrix operand order repaired and checked by reading, not compiled, not run on hardware
//
#include <hip/hip_runtime.h>
#include <hip/hip_bf16.h>
#include <stdint.h>

typedef __attribute__((ext_vector_type(16))) _Float16 v16h;
typedef __attribute__((ext_vector_type(8)))  float    v8f;

#define LSEQ   1024   // NUM_TILES^2
#define DMODEL 512
#define NHEAD  8
#define DHEAD  64
#define BATCH  4
#define BH     (BATCH * NHEAD)   // 32
#define NQ     (BH * LSEQ)       // 32768 (b,h,l) query slots

// CDNA5 async global->LDS path (ASYNCcnt-tracked, no VGPR staging).
// Flip to 0 to fall back to global_load_b128 + ds_store_b128.
#define USE_ASYNC_LDS 1

static __device__ __forceinline__ void async_load_b128(uint32_t lds_off,
                                                       const _Float16* gptr) {
  // LDS offset taken from the first VGPR operand; global addr from VGPR pair (GV mode)
  asm volatile("global_load_async_to_lds_b128 %0, %1, off"
               :: "v"(lds_off), "v"(gptr)
               : "memory");
}
static __device__ __forceinline__ void wait_async_le3() {
  asm volatile("s_wait_asynccnt 3" ::: "memory");
}
static __device__ __forceinline__ void wait_async_0() {
  asm volatile("s_wait_asynccnt 0" ::: "memory");
}
static __device__ __forceinline__ uint32_t lds_off_of(const void* p) {
  return (uint32_t)(uintptr_t)p;   // low 32 bits of flat shared address = LDS offset
}

static __device__ inline v8f zero8() {
  v8f z;
#pragma unroll
  for (int i = 0; i < 8; ++i) z[i] = 0.0f;
  return z;
}

// ---- WMMA fragment loaders (k = 16*(lane/16) + i in A and B, wave32) ----------

// A fragment: 16x32 f16 tile, src row-major [16][lda] (halves).
static __device__ inline v16h frag_a(const _Float16* src, int lda) {
  int lane = threadIdx.x & 31;
  const _Float16* p = src + (lane & 15) * lda + ((lane >> 4) * 16);
  v16h a;
#pragma unroll
  for (int i = 0; i < 16; ++i) a[i] = p[i];
  return a;
}

// B fragment: B[k][n] (32x16), column n stored as contiguous row n of src.
static __device__ inline v16h frag_b_nmajor(const _Float16* src, int ldb) {
  int lane = threadIdx.x & 31;
  const _Float16* p = src + (lane & 15) * ldb + ((lane >> 4) * 16);
  v16h b;
#pragma unroll
  for (int i = 0; i < 16; ++i) b[i] = p[i];
  return b;
}

// B fragment: B[k][n] stored (k,n) row-major [32][ldb].
static __device__ inline v16h frag_b_kmajor(const _Float16* src, int ldb) {
  int lane = threadIdx.x & 31;
  int n = lane & 15, kb = (lane >> 4) * 16;
  v16h b;
#pragma unroll
  for (int i = 0; i < 16; ++i) b[i] = src[(kb + i) * ldb + n];
  return b;
}

// ---- Kernel 1: f32 -> f16 convert ------------------------------------------
__global__ void f32_to_f16_kernel(const float* __restrict__ src,
                                  _Float16* __restrict__ dst, int n) {
  int i = blockIdx.x * blockDim.x + threadIdx.x;
  int stride = gridDim.x * blockDim.x;
  for (; i < n; i += stride) dst[i] = (_Float16)src[i];
}

// ---- Kernel 2: QKV projection GEMM (M=4096, N=512, K=512, z selects W) -----
// Block tile 128x64, 8 waves of 32x32, double-buffered async LDS staging.
__global__ __launch_bounds__(256)
void qkv_gemm_kernel(const _Float16* __restrict__ xh,
                     const _Float16* __restrict__ wh,    // [3][512][512]
                     _Float16* __restrict__ qkvh) {      // [3][BH][LSEQ][64]
  __shared__ _Float16 As[2][128 * 40];   // rows padded to 80B (16B-aligned b128)
  __shared__ _Float16 Bs[2][64 * 40];
  const int t    = threadIdx.x;
  const int wave = t >> 5;
  const int wr   = wave >> 1;          // 0..3 (32-row strip)
  const int wc   = wave & 1;           // 0..1 (32-col strip)
  const int lane = t & 31;
  const int n0   = lane & 15, m0 = (lane >> 4) * 8;

  const _Float16* Ag = xh + (size_t)blockIdx.x * 128 * DMODEL;
  const _Float16* Bg = wh + (size_t)blockIdx.z * DMODEL * DMODEL
                          + (size_t)blockIdx.y * 64 * DMODEL;
  _Float16* dst = qkvh + (size_t)blockIdx.z * ((size_t)NQ * DHEAD);

  v8f acc[2][2];
#pragma unroll
  for (int i = 0; i < 2; ++i)
#pragma unroll
    for (int j = 0; j < 2; ++j) acc[i][j] = zero8();

  const int arow = t >> 1, akoff = (t & 1) * 16;   // 16 halves/thread for A
  const int brow = t >> 2, bkoff = (t & 3) * 8;    //  8 halves/thread for B
  const _Float16* agp = Ag + arow * DMODEL + akoff;
  const _Float16* bgp = Bg + brow * DMODEL + bkoff;

#if USE_ASYNC_LDS
  // preload k-tile 0 into buffer 0 (3 async ops / thread, in-order completion)
  async_load_b128(lds_off_of(&As[0][arow * 40 + akoff]),      agp);
  async_load_b128(lds_off_of(&As[0][arow * 40 + akoff]) + 16, agp + 8);
  async_load_b128(lds_off_of(&Bs[0][brow * 40 + bkoff]),      bgp);
#endif

  int buf = 0;
  for (int kt = 0; kt < DMODEL; kt += 32) {
#if USE_ASYNC_LDS
    if (kt + 32 < DMODEL) {
      // issue next tile into the other buffer, then drain the current set
      async_load_b128(lds_off_of(&As[buf ^ 1][arow * 40 + akoff]),      agp + kt + 32);
      async_load_b128(lds_off_of(&As[buf ^ 1][arow * 40 + akoff]) + 16, agp + kt + 32 + 8);
      async_load_b128(lds_off_of(&Bs[buf ^ 1][brow * 40 + bkoff]),      bgp + kt + 32);
      __builtin_prefetch(agp + kt + 64, 0, 3);   // global_prefetch_b8, tile t+2
      wait_async_le3();
    } else {
      wait_async_0();
    }
    __syncthreads();
#else
    __syncthreads();
#pragma unroll
    for (int i = 0; i < 16; ++i)
      As[buf][arow * 40 + akoff + i] = agp[kt + i];
#pragma unroll
    for (int i = 0; i < 8; ++i)
      Bs[buf][brow * 40 + bkoff + i] = bgp[kt + i];
    __syncthreads();
#endif

    v16h a0 = frag_a(&As[buf][(wr * 32 + 0)  * 40], 40);
    v16h a1 = frag_a(&As[buf][(wr * 32 + 16) * 40], 40);
    v16h b0 = frag_b_nmajor(&Bs[buf][(wc * 32 + 0)  * 40], 40);
    v16h b1 = frag_b_nmajor(&Bs[buf][(wc * 32 + 16) * 40], 40);
    acc[0][0] = __builtin_amdgcn_wmma_f32_16x16x32_f16(false, a0, false, b0, (short)0, acc[0][0], false, false);
    acc[0][1] = __builtin_amdgcn_wmma_f32_16x16x32_f16(false, a0, false, b1, (short)0, acc[0][1], false, false);
    acc[1][0] = __builtin_amdgcn_wmma_f32_16x16x32_f16(false, a1, false, b0, (short)0, acc[1][0], false, false);
    acc[1][1] = __builtin_amdgcn_wmma_f32_16x16x32_f16(false, a1, false, b1, (short)0, acc[1][1], false, false);

#if USE_ASYNC_LDS
    __syncthreads();   // compute done before next preload overwrites this buffer
#endif
    buf ^= 1;
  }

#pragma unroll
  for (int mt = 0; mt < 2; ++mt)
#pragma unroll
    for (int nt = 0; nt < 2; ++nt)
#pragma unroll
      for (int v = 0; v < 8; ++v) {
        int gm = blockIdx.x * 128 + wr * 32 + mt * 16 + m0 + v;  // b*L + l
        int gn = blockIdx.y * 64  + wc * 32 + nt * 16 + n0;      // h*64 + d
        int b = gm >> 10, l = gm & 1023;
        int h = gn >> 6,  d = gn & 63;
        dst[((size_t)(b * NHEAD + h) * LSEQ + l) * DHEAD + d] = (_Float16)acc[mt][nt][v];
      }
}

// ---- Kernels 3/4: row-part / col-part attention ----------------------------
// mode 0: block = (bh, grid-row r). queries & keys are l = rc*32 + t.
// mode 1: block = (bh, grid-col c). queries & keys are l = t*32 + rc; the
//         duplicated self-key is the score-tile diagonal, zeroed here.
__global__ __launch_bounds__(32)
void attn_part_kernel(const _Float16* __restrict__ qh,
                      const _Float16* __restrict__ kh,
                      const _Float16* __restrict__ vh,
                      float* __restrict__ accP,   // [NQ][64]
                      float* __restrict__ mP,     // [NQ]
                      float* __restrict__ lP,     // [NQ]
                      int mode) {
  __shared__ _Float16 Qs[32 * 72];    // 144B row stride: 16B-aligned b128
  __shared__ _Float16 Ks[32 * 72];
  __shared__ _Float16 Vs[32 * 72];
  __shared__ float    Sc[32 * 33];
  __shared__ _Float16 Ps[32 * 33];

  const int bh   = blockIdx.x >> 5;
  const int rc   = blockIdx.x & 31;
  const int lane = threadIdx.x & 31;
  const size_t base = (size_t)bh * LSEQ * DHEAD;
  const int lt = (mode == 0) ? (rc * 32 + lane) : (lane * 32 + rc);

#if USE_ASYNC_LDS
  {
    const _Float16* qp = qh + base + (size_t)lt * DHEAD;
    const _Float16* kp = kh + base + (size_t)lt * DHEAD;
    const _Float16* vp = vh + base + (size_t)lt * DHEAD;
    uint32_t qo = lds_off_of(&Qs[lane * 72]);
    uint32_t ko = lds_off_of(&Ks[lane * 72]);
    uint32_t vo = lds_off_of(&Vs[lane * 72]);
#pragma unroll
    for (int c = 0; c < 4; ++c) {      // 4 x b128 = 32 halves per row
      async_load_b128(qo + c * 16, qp + c * 8);
      async_load_b128(ko + c * 16, kp + c * 8);
      async_load_b128(vo + c * 16, vp + c * 8);
    }
    wait_async_0();
  }
#else
#pragma unroll
  for (int i = 0; i < DHEAD; ++i) {
    Qs[lane * 72 + i] = qh[base + (size_t)lt * DHEAD + i];
    Ks[lane * 72 + i] = kh[base + (size_t)lt * DHEAD + i];
    Vs[lane * 72 + i] = vh[base + (size_t)lt * DHEAD + i];
  }
#endif
  __syncthreads();

  const float scale = 0.125f;   // 1/sqrt(64)
  const int n0 = lane & 15, m0 = (lane >> 4) * 8;

  // scores S = Q(32x64) @ K^T -> 32x32
#pragma unroll
  for (int mt = 0; mt < 2; ++mt)
#pragma unroll
    for (int nt = 0; nt < 2; ++nt) {
      v8f c = zero8();
#pragma unroll
      for (int ks = 0; ks < 2; ++ks) {
        v16h a = frag_a(&Qs[(mt * 16) * 72 + ks * 32], 72);
        v16h b = frag_b_nmajor(&Ks[(nt * 16) * 72 + ks * 32], 72);
        c = __builtin_amdgcn_wmma_f32_16x16x32_f16(false, a, false, b, (short)0, c, false, false);
      }
#pragma unroll
      for (int v = 0; v < 8; ++v)
        Sc[(mt * 16 + m0 + v) * 33 + nt * 16 + n0] = c[v] * scale;
    }
  __syncthreads();

  // per-query partial softmax (lane = local query index)
  {
    float mx = -3.0e38f;
#pragma unroll
    for (int j = 0; j < 32; ++j) mx = fmaxf(mx, Sc[lane * 33 + j]);
    float sum = 0.0f;
#pragma unroll
    for (int j = 0; j < 32; ++j) {
      float e = __expf(Sc[lane * 33 + j] - mx);
      if (mode == 1 && j == lane) e = 0.0f;   // drop duplicated self-key
      sum += e;
      Ps[lane * 33 + j] = (_Float16)e;
    }
    int gq = bh * LSEQ + lt;
    mP[gq] = mx;
    lP[gq] = sum;
  }
  __syncthreads();

  // acc = P(32x32) @ V(32x64)
#pragma unroll
  for (int mt = 0; mt < 2; ++mt) {
    v16h a = frag_a(&Ps[(mt * 16) * 33], 33);
#pragma unroll
    for (int nt = 0; nt < 4; ++nt) {
      v16h b = frag_b_kmajor(&Vs[nt * 16], 72);
      v8f c = zero8();
      c = __builtin_amdgcn_wmma_f32_16x16x32_f16(false, a, false, b, (short)0, c, false, false);
#pragma unroll
      for (int v = 0; v < 8; ++v) {
        int qloc = mt * 16 + m0 + v;
        int lq = (mode == 0) ? (rc * 32 + qloc) : (qloc * 32 + rc);
        accP[((size_t)bh * LSEQ + lq) * DHEAD + nt * 16 + n0] = c[v];
      }
    }
  }
}

// ---- Kernel 5: merge row/col flash partials, write (b,l,h*64+d) f32 --------
__global__ void attn_combine_kernel(const float* __restrict__ accR,
                                    const float* __restrict__ accC,
                                    const float* __restrict__ mR,
                                    const float* __restrict__ lR,
                                    const float* __restrict__ mC,
                                    const float* __restrict__ lC,
                                    float* __restrict__ out) {
  int idx = blockIdx.x * blockDim.x + threadIdx.x;
  if (idx >= NQ * DHEAD) return;
  int q = idx >> 6;          // (b,h,l) slot
  int d = idx & 63;
  int bh = q >> 10, l = q & 1023;
  int b = bh >> 3,  h = bh & 7;
  float m1 = mR[q], m2 = mC[q];
  float m  = fmaxf(m1, m2);
  float e1 = __expf(m1 - m), e2 = __expf(m2 - m);
  float denom = lR[q] * e1 + lC[q] * e2;
  float num   = accR[idx] * e1 + accC[idx] * e2;
  out[(size_t)(b * LSEQ + l) * DMODEL + h * DHEAD + d] = num / denom;
}

// ---- host ------------------------------------------------------------------
extern "C" void kernel_launch(void* const* d_in, const int* in_sizes, int n_in,
                              void* d_out, int out_size, void* d_ws, size_t ws_size,
                              hipStream_t stream) {
  const float* x  = (const float*)d_in[0];
  const float* Wq = (const float*)d_in[1];
  const float* Wk = (const float*)d_in[2];
  const float* Wv = (const float*)d_in[3];
  // d_in[4] (key_indices) unused: row/col tile structure derived analytically.

  char* ws = (char*)d_ws;
  size_t off = 0;
  _Float16* xh   = (_Float16*)(ws + off); off += (size_t)BATCH * LSEQ * DMODEL * 2;
  _Float16* wh   = (_Float16*)(ws + off); off += (size_t)3 * DMODEL * DMODEL * 2;
  _Float16* qkvh = (_Float16*)(ws + off); off += (size_t)3 * NQ * DHEAD * 2;
  float* accR = (float*)(ws + off); off += (size_t)NQ * DHEAD * 4;
  float* accC = (float*)(ws + off); off += (size_t)NQ * DHEAD * 4;
  float* mR = (float*)(ws + off); off += (size_t)NQ * 4;
  float* lR = (float*)(ws + off); off += (size_t)NQ * 4;
  float* mC = (float*)(ws + off); off += (size_t)NQ * 4;
  float* lC = (float*)(ws + off); off += (size_t)NQ * 4;
  (void)ws_size; (void)in_sizes; (void)n_in; (void)out_size;  // ~35.7 MB used

  const int nx = BATCH * LSEQ * DMODEL;
  const int nw = DMODEL * DMODEL;
  f32_to_f16_kernel<<<(nx + 255) / 256, 256, 0, stream>>>(x,  xh, nx);
  f32_to_f16_kernel<<<(nw + 255) / 256, 256, 0, stream>>>(Wq, wh + 0 * (size_t)nw, nw);
  f32_to_f16_kernel<<<(nw + 255) / 256, 256, 0, stream>>>(Wk, wh + 1 * (size_t)nw, nw);
  f32_to_f16_kernel<<<(nw + 255) / 256, 256, 0, stream>>>(Wv, wh + 2 * (size_t)nw, nw);

  dim3 ggrid(32, 8, 3);   // M/128, N/64, {Q,K,V}
  qkv_gemm_kernel<<<ggrid, 256, 0, stream>>>(xh, wh, qkvh);

  const _Float16* qh = qkvh + 0 * (size_t)NQ * DHEAD;
  const _Float16* kh = qkvh + 1 * (size_t)NQ * DHEAD;
  const _Float16* vh = qkvh + 2 * (size_t)NQ * DHEAD;
  attn_part_kernel<<<BH * 32, 32, 0, stream>>>(qh, kh, vh, accR, mR, lR, 0);
  attn_part_kernel<<<BH * 32, 32, 0, stream>>>(qh, kh, vh, accC, mC, lC, 1);

  attn_combine_kernel<<<(NQ * DHEAD + 255) / 256, 256, 0, stream>>>(
      accR, accC, mR, lR, mC, lC, (float*)d_out);
}
